// SwitchFFN_5952824673127
// MI455X (gfx1250) — matrix operand order repaired and checked by reading, not compiled, not run on hardware
//
#include <hip/hip_runtime.h>
#include <hip/hip_bf16.h>
#include <stdint.h>

// Problem constants (B=2, S=2048, D=768, F=3072, E=8)
#define T_TOK 4096
#define DMODEL 768
#define DFF 3072
#define NEXP 8

typedef __attribute__((ext_vector_type(16))) __bf16 bf16x16;
typedef __attribute__((ext_vector_type(8)))  float  f32x8;
typedef __attribute__((ext_vector_type(8)))  unsigned short u16x8;
typedef __attribute__((ext_vector_type(4)))  unsigned short u16x4;

struct U16x16 { u16x8 lo, hi; };

__device__ __forceinline__ unsigned short f2bf(float f) {
  // round-to-nearest-even fp32 -> bf16
  unsigned int u = __builtin_bit_cast(unsigned int, f);
  unsigned int r = u + 0x7FFFu + ((u >> 16) & 1u);
  return (unsigned short)(r >> 16);
}
__device__ __forceinline__ unsigned int pack2bf(float lo, float hi) {
  return (unsigned int)f2bf(lo) | ((unsigned int)f2bf(hi) << 16);
}
__device__ __forceinline__ bf16x16 mkfrag(u16x8 lo, u16x8 hi) {
  U16x16 t{lo, hi};
  return __builtin_bit_cast(bf16x16, t);
}

// ---------------------------------------------------------------- init
__global__ void moe_init(int* __restrict__ cnt, int* __restrict__ ctr) {
  int i = threadIdx.x;
  if (i < NEXP) { cnt[i] = 0; ctr[i] = 0; }
}

// ---------------------------------------------------------------- router
__global__ __launch_bounds__(256)
void moe_router(const float* __restrict__ x, const float* __restrict__ wsw,
                const float* __restrict__ bsw, int* __restrict__ routes,
                int* __restrict__ cnt, float* __restrict__ prob) {
  int gid = blockIdx.x * 256 + threadIdx.x;
  int t = gid >> 5;
  int lane = gid & 31;
  if (t >= T_TOK) return;

  float acc[NEXP];
#pragma unroll
  for (int e = 0; e < NEXP; ++e) acc[e] = 0.f;

  const float* xr = x + (size_t)t * DMODEL;
  for (int k = lane; k < DMODEL; k += 32) {
    float xv = xr[k];
#pragma unroll
    for (int e = 0; e < NEXP; ++e) acc[e] += xv * wsw[k * NEXP + e];
  }
#pragma unroll
  for (int e = 0; e < NEXP; ++e) {
#pragma unroll
    for (int s = 16; s > 0; s >>= 1) acc[e] += __shfl_xor(acc[e], s, 32);
  }
  if (lane == 0) {
    float lg[NEXP];
    float mx = acc[0] + bsw[0]; int arg = 0;
    lg[0] = mx;
#pragma unroll
    for (int e = 1; e < NEXP; ++e) {
      lg[e] = acc[e] + bsw[e];
      if (lg[e] > mx) { mx = lg[e]; arg = e; }   // first-max (argmax semantics)
    }
    float s = 0.f, p[NEXP];
#pragma unroll
    for (int e = 0; e < NEXP; ++e) { p[e] = __expf(lg[e] - mx); s += p[e]; }
    float inv = 1.0f / s;
#pragma unroll
    for (int e = 0; e < NEXP; ++e) prob[(size_t)t * NEXP + e] = p[e] * inv;
    routes[t] = arg;
    atomicAdd(&cnt[arg], 1);
  }
}

// ---------------------------------------------------------------- finalize router
__global__ __launch_bounds__(256)
void moe_finalize(const int* __restrict__ cnt, int* __restrict__ off,
                  const float* __restrict__ prob, float* __restrict__ out_tail) {
  __shared__ float red[256];
  int tid = threadIdx.x;
  if (tid == 0) {
    int o = 0;
    for (int e = 0; e < NEXP; ++e) { off[e] = o; o += cnt[e]; }
    out_tail[16] = 0.0f;               // n_dropped = 0
  }
  if (tid < NEXP) out_tail[tid] = (float)cnt[tid];
  for (int e = 0; e < NEXP; ++e) {
    float s = 0.f;
    for (int t = tid; t < T_TOK; t += 256) s += prob[(size_t)t * NEXP + e];
    red[tid] = s;
    __syncthreads();
    for (int st = 128; st > 0; st >>= 1) {
      if (tid < st) red[tid] += red[tid + st];
      __syncthreads();
    }
    if (tid == 0) out_tail[8 + e] = red[0];
    __syncthreads();
  }
}

// ---------------------------------------------------------------- scatter
__global__ __launch_bounds__(256)
void moe_scatter(const int* __restrict__ routes, const int* __restrict__ off,
                 int* __restrict__ ctr, int* __restrict__ list) {
  int t = blockIdx.x * 256 + threadIdx.x;
  if (t >= T_TOK) return;
  int e = routes[t];
  int pos = atomicAdd(&ctr[e], 1);
  list[off[e] + pos] = t;
}

// LDS buffer strides in u16x8 (16B) units for ping-pong indexing
#define A_BUF_STRIDE8 (32 * 40 / 8)     // 160
#define B_BUF_STRIDE8 (128 * 40 / 8)    // 640

// ---------------------------------------------------------------- grouped GEMM 1
// H[base+m, n] = relu( sum_k x[list[m], k] * wi[e][k, n] )   (K=768, N=3072)
// block tile: M=32 x N=128, 8 waves, each wave -> 16x32, double-buffered LDS
__global__ __launch_bounds__(256)
void moe_ffn1(const float* __restrict__ x, const float* __restrict__ wi,
              const int* __restrict__ cnt, const int* __restrict__ off,
              const int* __restrict__ list, unsigned short* __restrict__ H) {
  const int e = blockIdx.z;
  const int Ne = cnt[e];
  const int m0 = blockIdx.x * 32;
  if (m0 >= Ne) return;
  const int n0 = blockIdx.y * 128;
  const int base = off[e];

  __shared__ __align__(16) unsigned short As[2][32][40];    // [buf][m][k]
  __shared__ __align__(16) unsigned short Bs[2][128][40];   // [buf][n][k]
  __shared__ int tokrow[32];

  const int tid = threadIdx.x;
  const int lane = tid & 31;
  const int w = tid >> 5;
  const int wm = (w & 1) << 4;
  const int wn = (w >> 1) << 5;
  const int half = lane >> 4;
  const int l15 = lane & 15;

  if (tid < 32) {
    int gm = m0 + tid;
    tokrow[tid] = (gm < Ne) ? list[base + gm] : -1;
  }
  __syncthreads();

  // A loader: one float4 per thread per K-step (loop-invariant row/token)
  const int ar_row = tid >> 3;
  const int ar_col = (tid & 7) << 2;
  const int atok = tokrow[ar_row];
  const float* aptr = x + (size_t)(atok < 0 ? 0 : atok) * DMODEL + ar_col;

  const float* wiE = wi + (size_t)e * DMODEL * DFF + n0;

  // fragment LDS pointers (buffer 0); add p*stride for ping-pong
  const u16x8* fa0  = (const u16x8*)&As[0][wm + l15][half * 16];
  const u16x8* fa1  = (const u16x8*)&As[0][wm + l15][half * 16 + 8];
  const u16x8* fb00 = (const u16x8*)&Bs[0][wn + l15][half * 16];
  const u16x8* fb01 = (const u16x8*)&Bs[0][wn + l15][half * 16 + 8];
  const u16x8* fb10 = (const u16x8*)&Bs[0][wn + 16 + l15][half * 16];
  const u16x8* fb11 = (const u16x8*)&Bs[0][wn + 16 + l15][half * 16 + 8];

  f32x8 acc0 = {}, acc1 = {};

  // ---- prologue: stage K-step 0 into buffer 0 ----
  {
    float4 av = *(const float4*)(aptr + 0);
    if (atok < 0) { av.x = 0.f; av.y = 0.f; av.z = 0.f; av.w = 0.f; }
    float b0[8], b1[8];
#pragma unroll
    for (int j = 0; j < 8; ++j) {
      int idx = tid + j * 256;
      int nn = idx & 127, kp = idx >> 7;
      const float* pb = wiE + (size_t)(2 * kp) * DFF + nn;
      b0[j] = pb[0];
      b1[j] = pb[DFF];
    }
    *(uint2*)&As[0][ar_row][ar_col] = make_uint2(pack2bf(av.x, av.y), pack2bf(av.z, av.w));
#pragma unroll
    for (int j = 0; j < 8; ++j) {
      int idx = tid + j * 256;
      int nn = idx & 127, kp = idx >> 7;
      *(unsigned int*)&Bs[0][nn][kp * 2] = pack2bf(b0[j], b1[j]);
    }
  }
  __syncthreads();

  int p = 0;
  for (int k0 = 32; k0 < DMODEL; k0 += 32) {
    // global loads for next step (batched, overlap with compute)
    float4 av = *(const float4*)(aptr + k0);
    if (atok < 0) { av.x = 0.f; av.y = 0.f; av.z = 0.f; av.w = 0.f; }
    float b0[8], b1[8];
#pragma unroll
    for (int j = 0; j < 8; ++j) {
      int idx = tid + j * 256;
      int nn = idx & 127, kp = idx >> 7;
      const float* pb = wiE + (size_t)(k0 + 2 * kp) * DFF + nn;
      b0[j] = pb[0];
      b1[j] = pb[DFF];
    }

    // compute current step from buffer p
    {
      int o = p * A_BUF_STRIDE8, ob = p * B_BUF_STRIDE8;
      bf16x16 a  = mkfrag(fa0[o], fa1[o]);
      bf16x16 bA = mkfrag(fb00[ob], fb01[ob]);
      bf16x16 bB = mkfrag(fb10[ob], fb11[ob]);
      acc0 = __builtin_amdgcn_wmma_f32_16x16x32_bf16(false, a, false, bA,
                                                     (short)0, acc0, false, false);
      acc1 = __builtin_amdgcn_wmma_f32_16x16x32_bf16(false, a, false, bB,
                                                     (short)0, acc1, false, false);
    }

    // stage next step into buffer 1-p
    int q = p ^ 1;
    *(uint2*)&As[q][ar_row][ar_col] = make_uint2(pack2bf(av.x, av.y), pack2bf(av.z, av.w));
#pragma unroll
    for (int j = 0; j < 8; ++j) {
      int idx = tid + j * 256;
      int nn = idx & 127, kp = idx >> 7;
      *(unsigned int*)&Bs[q][nn][kp * 2] = pack2bf(b0[j], b1[j]);
    }
    __syncthreads();
    p = q;
  }

  // ---- epilogue compute ----
  {
    int o = p * A_BUF_STRIDE8, ob = p * B_BUF_STRIDE8;
    bf16x16 a  = mkfrag(fa0[o], fa1[o]);
    bf16x16 bA = mkfrag(fb00[ob], fb01[ob]);
    bf16x16 bB = mkfrag(fb10[ob], fb11[ob]);
    acc0 = __builtin_amdgcn_wmma_f32_16x16x32_bf16(false, a, false, bA,
                                                   (short)0, acc0, false, false);
    acc1 = __builtin_amdgcn_wmma_f32_16x16x32_bf16(false, a, false, bB,
                                                   (short)0, acc1, false, false);
  }

  const int gn0 = n0 + wn + l15;
#pragma unroll
  for (int i = 0; i < 8; ++i) {
    int lm = wm + half * 8 + i;             // C/D layout: VGPR i -> M = i + 8*half
    if (m0 + lm < Ne) {
      size_t row = (size_t)(base + m0 + lm) * DFF;
      float v0 = acc0[i]; v0 = v0 > 0.f ? v0 : 0.f;
      float v1 = acc1[i]; v1 = v1 > 0.f ? v1 : 0.f;
      H[row + gn0]      = f2bf(v0);
      H[row + gn0 + 16] = f2bf(v1);
    }
  }
}

// ---------------------------------------------------------------- grouped GEMM 2
// out[list[m], n] = sum_k H[base+m, k] * wo[e][k, n]   (K=3072, N=768)
__global__ __launch_bounds__(256)
void moe_ffn2(const unsigned short* __restrict__ H, const float* __restrict__ wo,
              const int* __restrict__ cnt, const int* __restrict__ off,
              const int* __restrict__ list, float* __restrict__ out) {
  const int e = blockIdx.z;
  const int Ne = cnt[e];
  const int m0 = blockIdx.x * 32;
  if (m0 >= Ne) return;
  const int n0 = blockIdx.y * 128;
  const int base = off[e];

  __shared__ __align__(16) unsigned short As[2][32][40];
  __shared__ __align__(16) unsigned short Bs[2][128][40];
  __shared__ int tokrow[32];

  const int tid = threadIdx.x;
  const int lane = tid & 31;
  const int w = tid >> 5;
  const int wm = (w & 1) << 4;
  const int wn = (w >> 1) << 5;
  const int half = lane >> 4;
  const int l15 = lane & 15;

  if (tid < 32) {
    int gm = m0 + tid;
    tokrow[tid] = (gm < Ne) ? list[base + gm] : -1;
  }
  __syncthreads();

  // A loader: 4 contiguous bf16 (8B) per thread per K-step
  const int ar_row = tid >> 3;
  const int ar_col = (tid & 7) << 2;
  const bool avalid = (m0 + ar_row) < Ne;
  const unsigned short* aptr =
      H + (size_t)(base + m0 + (avalid ? ar_row : 0)) * DFF + ar_col;

  const float* woE = wo + (size_t)e * DFF * DMODEL + n0;

  const u16x8* fa0  = (const u16x8*)&As[0][wm + l15][half * 16];
  const u16x8* fa1  = (const u16x8*)&As[0][wm + l15][half * 16 + 8];
  const u16x8* fb00 = (const u16x8*)&Bs[0][wn + l15][half * 16];
  const u16x8* fb01 = (const u16x8*)&Bs[0][wn + l15][half * 16 + 8];
  const u16x8* fb10 = (const u16x8*)&Bs[0][wn + 16 + l15][half * 16];
  const u16x8* fb11 = (const u16x8*)&Bs[0][wn + 16 + l15][half * 16 + 8];

  f32x8 acc0 = {}, acc1 = {};

  // ---- prologue: stage K-step 0 into buffer 0 ----
  {
    u16x4 av = *(const u16x4*)(aptr + 0);
    if (!avalid) { av.x = 0; av.y = 0; av.z = 0; av.w = 0; }
    float b0[8], b1[8];
#pragma unroll
    for (int j = 0; j < 8; ++j) {
      int idx = tid + j * 256;
      int nn = idx & 127, kp = idx >> 7;
      const float* pb = woE + (size_t)(2 * kp) * DMODEL + nn;
      b0[j] = pb[0];
      b1[j] = pb[DMODEL];
    }
    *(u16x4*)&As[0][ar_row][ar_col] = av;
#pragma unroll
    for (int j = 0; j < 8; ++j) {
      int idx = tid + j * 256;
      int nn = idx & 127, kp = idx >> 7;
      *(unsigned int*)&Bs[0][nn][kp * 2] = pack2bf(b0[j], b1[j]);
    }
  }
  __syncthreads();

  int p = 0;
  for (int k0 = 32; k0 < DFF; k0 += 32) {
    u16x4 av = *(const u16x4*)(aptr + k0);
    if (!avalid) { av.x = 0; av.y = 0; av.z = 0; av.w = 0; }
    float b0[8], b1[8];
#pragma unroll
    for (int j = 0; j < 8; ++j) {
      int idx = tid + j * 256;
      int nn = idx & 127, kp = idx >> 7;
      const float* pb = woE + (size_t)(k0 + 2 * kp) * DMODEL + nn;
      b0[j] = pb[0];
      b1[j] = pb[DMODEL];
    }

    {
      int o = p * A_BUF_STRIDE8, ob = p * B_BUF_STRIDE8;
      bf16x16 a  = mkfrag(fa0[o], fa1[o]);
      bf16x16 bA = mkfrag(fb00[ob], fb01[ob]);
      bf16x16 bB = mkfrag(fb10[ob], fb11[ob]);
      acc0 = __builtin_amdgcn_wmma_f32_16x16x32_bf16(false, a, false, bA,
                                                     (short)0, acc0, false, false);
      acc1 = __builtin_amdgcn_wmma_f32_16x16x32_bf16(false, a, false, bB,
                                                     (short)0, acc1, false, false);
    }

    int q = p ^ 1;
    *(u16x4*)&As[q][ar_row][ar_col] = av;
#pragma unroll
    for (int j = 0; j < 8; ++j) {
      int idx = tid + j * 256;
      int nn = idx & 127, kp = idx >> 7;
      *(unsigned int*)&Bs[q][nn][kp * 2] = pack2bf(b0[j], b1[j]);
    }
    __syncthreads();
    p = q;
  }

  {
    int o = p * A_BUF_STRIDE8, ob = p * B_BUF_STRIDE8;
    bf16x16 a  = mkfrag(fa0[o], fa1[o]);
    bf16x16 bA = mkfrag(fb00[ob], fb01[ob]);
    bf16x16 bB = mkfrag(fb10[ob], fb11[ob]);
    acc0 = __builtin_amdgcn_wmma_f32_16x16x32_bf16(false, a, false, bA,
                                                   (short)0, acc0, false, false);
    acc1 = __builtin_amdgcn_wmma_f32_16x16x32_bf16(false, a, false, bB,
                                                   (short)0, acc1, false, false);
  }

  const int gn0 = n0 + wn + l15;
#pragma unroll
  for (int i = 0; i < 8; ++i) {
    int lm = wm + half * 8 + i;
    int t = tokrow[lm];
    if (t >= 0) {
      out[(size_t)t * DMODEL + gn0]      = acc0[i];
      out[(size_t)t * DMODEL + gn0 + 16] = acc1[i];
    }
  }
}

// ---------------------------------------------------------------- launch
extern "C" void kernel_launch(void* const* d_in, const int* in_sizes, int n_in,
                              void* d_out, int out_size, void* d_ws, size_t ws_size,
                              hipStream_t stream) {
  const float* x   = (const float*)d_in[0];   // [T, D]
  const float* wsw = (const float*)d_in[1];   // [D, E]
  const float* bsw = (const float*)d_in[2];   // [E]
  const float* wi  = (const float*)d_in[3];   // [E, D, F]
  const float* wo  = (const float*)d_in[4];   // [E, F, D]
  float* out = (float*)d_out;
  float* out_tail = out + (size_t)T_TOK * DMODEL;   // counts[8], probsum[8], n_dropped

  char* p = (char*)d_ws;
  int* routes = (int*)p;           p += (size_t)T_TOK * sizeof(int);
  int* cnt    = (int*)p;           p += 16 * sizeof(int);
  int* ctr    = (int*)p;           p += 16 * sizeof(int);
  int* off    = (int*)p;           p += 16 * sizeof(int);
  float* prob = (float*)p;         p += (size_t)T_TOK * NEXP * sizeof(float);
  int* list   = (int*)p;           p += (size_t)T_TOK * sizeof(int);
  p = (char*)(((uintptr_t)p + 255) & ~(uintptr_t)255);
  unsigned short* H = (unsigned short*)p;     // [T, F] bf16 (~25 MB)

  moe_init    <<<1, 32, 0, stream>>>(cnt, ctr);
  moe_router  <<<T_TOK / 8, 256, 0, stream>>>(x, wsw, bsw, routes, cnt, prob);
  moe_finalize<<<1, 256, 0, stream>>>(cnt, off, prob, out_tail);
  moe_scatter <<<T_TOK / 256, 256, 0, stream>>>(routes, off, ctr, list);
  moe_ffn1<<<dim3(T_TOK / 32, DFF / 128, NEXP), 256, 0, stream>>>(x, wi, cnt, off, list, H);
  moe_ffn2<<<dim3(T_TOK / 32, DMODEL / 128, NEXP), 256, 0, stream>>>(H, wo, cnt, off, list, out);
}
